// SSM_40553081209591
// MI455X (gfx1250) — compile-verified, matrix operand order checked by reading, not run on hardware
//
#include <hip/hip_runtime.h>
#include <hip/hip_bf16.h>
#include <math.h>
#include <stdint.h>

// ---------------------------------------------------------------------------
// S5-style SSM forward, chunk-parallel 3-phase:
//   phase1: per (b,h,seg): Bu-GEMM + local scan from 0 -> segment-final state
//   phase2: serial combine over segments -> carry-in per segment
//   phase3: per (b,h,seg): Bu-GEMM + scan (with carry) + C-GEMM + D*u -> y
// Shapes: C_=1, H=8, N=64, Din=M=128, L=4096, B=16, SEGS=8 (512 steps each)
// Both einsums run as v_wmma_f32_16x16x32_f16; scan state stays in VGPRs.
// u staging uses gfx1250 global_load_async_to_lds_b128 (ASYNCcnt).
// ---------------------------------------------------------------------------

typedef __attribute__((ext_vector_type(16))) _Float16 v16h;
typedef __attribute__((ext_vector_type(8)))  _Float16 v8h;
typedef __attribute__((ext_vector_type(8)))  float    v8f;
typedef __attribute__((ext_vector_type(4)))  float    v4f;
typedef __attribute__((ext_vector_type(4)))  int      v4i;

// async-to-LDS pointer types (AS1 = global, AS3 = LDS), per clang's signature:
//   void __builtin_amdgcn_global_load_async_to_lds_b128(v4i AS1*, v4i AS3*, Ii, Ii)
typedef __attribute__((address_space(1))) v4i* gv4i_p;
typedef __attribute__((address_space(3))) v4i* lv4i_p;

#define SSM_H    8
#define SSM_N    64
#define SSM_D    128     // Din == output model dim M
#define SSM_L    4096
#define SSM_B    16
#define SSM_SEGS 8
#define SSM_SEG  (SSM_L / SSM_SEGS)   // 512 timesteps per segment
#define SSM_T    32                   // timesteps per chunk (2 WMMA M-tiles)

#if defined(__HIP_DEVICE_COMPILE__) && defined(__has_builtin)
#if __has_builtin(__builtin_amdgcn_global_load_async_to_lds_b128) && \
    __has_builtin(__builtin_amdgcn_s_wait_asynccnt)
#define SSM_HAS_ASYNC 1
#endif
#endif
#ifndef SSM_HAS_ASYNC
#define SSM_HAS_ASYNC 0
#endif

__device__ __forceinline__ float softplusf_dev(float x) {
    return (x > 20.f) ? x : log1pf(expf(x));
}

__device__ __forceinline__ v16h combine16(v8h a, v8h b) {
    v16h r;
#pragma unroll
    for (int j = 0; j < 8; ++j) { r[j] = a[j]; r[8 + j] = b[j]; }
    return r;
}

// --------------------------------------------------------------------------
// Precompute A_bar and A_bar^SEG (complex, per h,n)
// --------------------------------------------------------------------------
__global__ void ssm_precompA(const float* __restrict__ Lre,
                             const float* __restrict__ Lim,
                             const float* __restrict__ logdt,
                             float* __restrict__ Are,  float* __restrict__ Aim,
                             float* __restrict__ Pre,  float* __restrict__ Pim) {
    int id = blockIdx.x * blockDim.x + threadIdx.x;
    if (id >= SSM_H * SSM_N) return;
    float lre = -softplusf_dev(Lre[id]);
    float lim = Lim[id];
    float dt  = expf(logdt[id]);
    float e   = expf(lre * dt);
    Are[id] = e * cosf(lim * dt);
    Aim[id] = e * sinf(lim * dt);
    // A_bar^SEG = exp(SEG * Lambda_bar)
    float ep = expf((float)SSM_SEG * lre * dt);
    float th = (float)SSM_SEG * lim * dt;
    Pre[id] = ep * cosf(th);
    Pim[id] = ep * sinf(th);
}

// --------------------------------------------------------------------------
// Precompute WMMA B-operand fragments in the hardware register layout
// (ISA 7.12.2, 16-bit B matrix, wave32):
//   lane = (col & 15) + 16*khalf ;  element e = g*8 + j
//   K    = ks*32 + (g*2 + khalf)*8 + j
// Wb: K = d, cols = [Re(B_bar) n | Im(B_bar) n] ; Wc: K = [xre|xim], cols = m
// Memory layout: [h][coltile(8)][ks(4)][lane(32)][e(16)] halves
// --------------------------------------------------------------------------
__global__ void ssm_precompW(const float* __restrict__ Lre,
                             const float* __restrict__ Lim,
                             const float* __restrict__ logdt,
                             const float* __restrict__ Bre,
                             const float* __restrict__ Bim,
                             const float* __restrict__ Cre,
                             const float* __restrict__ Cim,
                             _Float16* __restrict__ Wb,
                             _Float16* __restrict__ Wc) {
    const int per = SSM_H * 128 * 128;      // 131072 elements each
    int id = blockIdx.x * blockDim.x + threadIdx.x;
    if (id >= 2 * per) return;
    bool isC = (id >= per);
    int e0  = id % per;
    int h   = e0 / (128 * 128);
    int rem = e0 % (128 * 128);

    if (!isC) {
        int c = rem / 128;       // packed output column: 0..63 re, 64..127 im
        int d = rem % 128;       // K index (model dim)
        int n = c & 63;
        int hn = h * SSM_N + n;
        float lre = -softplusf_dev(Lre[hn]);
        float lim = Lim[hn];
        float dt  = expf(logdt[hn]);
        float eab = expf(lre * dt);
        float are = eab * cosf(lim * dt);
        float aim = eab * sinf(lim * dt);
        float den = lre * lre + lim * lim + 1e-7f;
        float rre =  lre / den;
        float rim = -lim / den;
        float tre = are - 1.f, tim = aim;
        float bcre = rre * tre - rim * tim;
        float bcim = rre * tim + rim * tre;
        float br = Bre[hn * SSM_D + d];
        float bi = Bim[hn * SSM_D + d];
        float bbre = bcre * br - bcim * bi;
        float bbim = bcre * bi + bcim * br;
        float val = (c >= 64) ? bbim : bbre;

        int ct = c >> 4, cl = c & 15;
        int ks = d >> 5, r = d & 31;
        int g = r >> 4, khalf = (r >> 3) & 1, j = r & 7;
        Wb[(((h * 8 + ct) * 4 + ks) * 32 + (cl + (khalf << 4))) * 16 + g * 8 + j]
            = (_Float16)val;
    } else {
        int m = rem / 128;       // output column (model dim)
        int k = rem % 128;       // K index over packed [xre | xim]
        float val = (k < 64) ? ( 2.f * Cre[(h * 128 + m) * 64 + k])
                             : (-2.f * Cim[(h * 128 + m) * 64 + (k - 64)]);
        int ct = m >> 4, cl = m & 15;
        int ks = k >> 5, r = k & 31;
        int g = r >> 4, khalf = (r >> 3) & 1, j = r & 7;
        Wc[(((h * 8 + ct) * 4 + ks) * 32 + (cl + (khalf << 4))) * 16 + g * 8 + j]
            = (_Float16)val;
    }
}

// --------------------------------------------------------------------------
// Segment body. FULL=false: phase1 (summary only). FULL=true: phase3.
// One workgroup per (b,h,seg); 8 waves; chunk = 32 timesteps.
// --------------------------------------------------------------------------
template <bool FULL>
__global__ __launch_bounds__(256) void ssm_seg_kernel(
        const float* __restrict__ u,
        const float* __restrict__ Dvec,
        const float* __restrict__ Are,  const float* __restrict__ Aim,
        const _Float16* __restrict__ Wb, const _Float16* __restrict__ Wc,
        const float* __restrict__ carryRe, const float* __restrict__ carryIm,
        float* __restrict__ finRe, float* __restrict__ finIm,
        float* __restrict__ y) {
    const int blk = blockIdx.x;
    const int b   = blk / (SSM_H * SSM_SEGS);
    const int h   = (blk / SSM_SEGS) % SSM_H;
    const int s   = blk % SSM_SEGS;
    const int tid  = threadIdx.x;
    const int w    = tid >> 5;           // wave id 0..7 (wave32)
    const int lane = tid & 31;

    __shared__ __align__(16) float    u32_sh[SSM_T][128];  // u chunk f32 (16 KB)
    __shared__ __align__(16) _Float16 u16_sh[SSM_T][128];  // u chunk f16 ( 8 KB)
    __shared__ __align__(16) float    bu_sh [SSM_T][128];  // Bu / y tile (16 KB)
    __shared__ __align__(16) _Float16 x_sh  [SSM_T][128];  // x tile f16  ( 8 KB)

    // Resident weight fragments for this wave's 16-column tile.
    v16h Bfrag[4], Cfrag[4];
    {
        const _Float16* pb = Wb + ((size_t)((h * 8 + w) * 4) * 32 + lane) * 16;
#pragma unroll
        for (int ks = 0; ks < 4; ++ks)
            Bfrag[ks] = *(const v16h*)(pb + (size_t)ks * 32 * 16);
        if (FULL) {
            const _Float16* pc = Wc + ((size_t)((h * 8 + w) * 4) * 32 + lane) * 16;
#pragma unroll
            for (int ks = 0; ks < 4; ++ks)
                Cfrag[ks] = *(const v16h*)(pc + (size_t)ks * 32 * 16);
        }
    }

    // Staging geometry: thread owns row rr, 16 columns starting at cc.
    const int rr = tid >> 3;            // 0..31
    const int cc = (tid & 7) * 16;      // 0,16,...,112
    v4f dreg[4];
    if (FULL) {
#pragma unroll
        for (int q = 0; q < 4; ++q)
            dreg[q] = *(const v4f*)(Dvec + h * SSM_D + cc + q * 4);
    }

    // Recurrence carry: one complex state per lane (lanes 0..63).
    const int segbase = ((b * SSM_H + h) * SSM_SEGS + s) * SSM_N;
    float a_re = 0.f, a_im = 0.f, xre = 0.f, xim = 0.f;
    if (tid < SSM_N) {
        a_re = Are[h * SSM_N + tid];
        a_im = Aim[h * SSM_N + tid];
        if (FULL) { xre = carryRe[segbase + tid]; xim = carryIm[segbase + tid]; }
    }

    const int rowA  = lane & 15;   // A-fragment row within M-tile
    const int khalf = lane >> 4;
    const int t0acc = khalf * 8;   // acc VGPR r -> timestep (mt*16 + t0acc + r)
    const int mcol  = w * 16 + (lane & 15);
    const float* useg = u + ((size_t)b * SSM_L + s * SSM_SEG) * SSM_D;

    for (int chunk = 0; chunk < SSM_SEG / SSM_T; ++chunk) {
        const int l0 = chunk * SSM_T;
        const float* gp = useg + (size_t)(l0 + rr) * SSM_D + cc;

        // ---- stage u chunk: global -> u32_sh (f32) and u16_sh (f16) ----
#if SSM_HAS_ASYNC
#pragma unroll
        for (int q = 0; q < 4; ++q) {
            __builtin_amdgcn_global_load_async_to_lds_b128(
                (gv4i_p)(uintptr_t)(gp + q * 4),
                (lv4i_p)(uint32_t)(uintptr_t)(&u32_sh[rr][cc + q * 4]),
                0, 0);
        }
        if (chunk + 1 < SSM_SEG / SSM_T)
            __builtin_prefetch(gp + SSM_T * SSM_D, 0, 1);
        __builtin_amdgcn_s_wait_asynccnt(0);
        __syncthreads();
        {
            v4f q0 = *(const v4f*)&u32_sh[rr][cc];
            v4f q1 = *(const v4f*)&u32_sh[rr][cc + 4];
            v4f q2 = *(const v4f*)&u32_sh[rr][cc + 8];
            v4f q3 = *(const v4f*)&u32_sh[rr][cc + 12];
            v8h h0, h1;
#pragma unroll
            for (int j = 0; j < 4; ++j) {
                h0[j] = (_Float16)q0[j]; h0[4 + j] = (_Float16)q1[j];
                h1[j] = (_Float16)q2[j]; h1[4 + j] = (_Float16)q3[j];
            }
            *(v8h*)&u16_sh[rr][cc]     = h0;
            *(v8h*)&u16_sh[rr][cc + 8] = h1;
        }
        __syncthreads();
#else
        {
            v4f q0 = *(const v4f*)(gp);
            v4f q1 = *(const v4f*)(gp + 4);
            v4f q2 = *(const v4f*)(gp + 8);
            v4f q3 = *(const v4f*)(gp + 12);
            if (chunk + 1 < SSM_SEG / SSM_T)
                __builtin_prefetch(gp + SSM_T * SSM_D, 0, 1);
            *(v4f*)&u32_sh[rr][cc]      = q0;
            *(v4f*)&u32_sh[rr][cc + 4]  = q1;
            *(v4f*)&u32_sh[rr][cc + 8]  = q2;
            *(v4f*)&u32_sh[rr][cc + 12] = q3;
            v8h h0, h1;
#pragma unroll
            for (int j = 0; j < 4; ++j) {
                h0[j] = (_Float16)q0[j]; h0[4 + j] = (_Float16)q1[j];
                h1[j] = (_Float16)q2[j]; h1[4 + j] = (_Float16)q3[j];
            }
            *(v8h*)&u16_sh[rr][cc]     = h0;
            *(v8h*)&u16_sh[rr][cc + 8] = h1;
        }
        __syncthreads();
#endif

        // ---- GEMM1: Bu(32t x 128) = u16 * Wb  (2 M-tiles per wave) ----
        v8f acc[2] = {{}, {}};
#pragma unroll
        for (int mt = 0; mt < 2; ++mt) {
#pragma unroll
            for (int ks = 0; ks < 4; ++ks) {
                const _Float16* s0 = &u16_sh[mt * 16 + rowA][ks * 32 + khalf * 8];
                v16h af = combine16(*(const v8h*)s0, *(const v8h*)(s0 + 16));
                acc[mt] = __builtin_amdgcn_wmma_f32_16x16x32_f16(
                    false, af, false, Bfrag[ks], (short)0, acc[mt], false, false);
            }
        }
#pragma unroll
        for (int mt = 0; mt < 2; ++mt)
#pragma unroll
            for (int r = 0; r < 8; ++r)
                bu_sh[mt * 16 + t0acc + r][mcol] = acc[mt][r];
        __syncthreads();

        // ---- recurrence: x_t = A*x_{t-1} + Bu_t (one lane per state) ----
        if (tid < SSM_N) {
#pragma unroll
            for (int t = 0; t < SSM_T; ++t) {
                float bre = bu_sh[t][tid];
                float bim = bu_sh[t][tid + SSM_N];
                float nre = a_re * xre - a_im * xim + bre;
                float nim = a_re * xim + a_im * xre + bim;
                xre = nre; xim = nim;
                if (FULL) {
                    x_sh[t][tid]         = (_Float16)nre;
                    x_sh[t][tid + SSM_N] = (_Float16)nim;
                }
            }
        }
        __syncthreads();

        if (FULL) {
            // ---- GEMM2: y(32t x 128) = x * Wc, staged back into bu_sh ----
            v8f acc2[2] = {{}, {}};
#pragma unroll
            for (int mt = 0; mt < 2; ++mt) {
#pragma unroll
                for (int ks = 0; ks < 4; ++ks) {
                    const _Float16* s0 = &x_sh[mt * 16 + rowA][ks * 32 + khalf * 8];
                    v16h af = combine16(*(const v8h*)s0, *(const v8h*)(s0 + 16));
                    acc2[mt] = __builtin_amdgcn_wmma_f32_16x16x32_f16(
                        false, af, false, Cfrag[ks], (short)0, acc2[mt], false, false);
                }
            }
#pragma unroll
            for (int mt = 0; mt < 2; ++mt)
#pragma unroll
                for (int r = 0; r < 8; ++r)
                    bu_sh[mt * 16 + t0acc + r][mcol] = acc2[mt][r];
            __syncthreads();

            // ---- coalesced store: y = tile + D*u (b128 rows) ----
            float* yrow = y + (((size_t)b * SSM_L + s * SSM_SEG + l0 + rr) * SSM_H
                               + h) * SSM_D + cc;
#pragma unroll
            for (int q = 0; q < 4; ++q) {
                v4f t  = *(const v4f*)&bu_sh[rr][cc + q * 4];
                v4f uu = *(const v4f*)&u32_sh[rr][cc + q * 4];
                *(v4f*)(yrow + q * 4) = t + dreg[q] * uu;
            }
        }
        __syncthreads();   // protect LDS tiles before next chunk overwrites
    }

    if (!FULL) {
        if (tid < SSM_N) {
            finRe[segbase + tid] = xre;
            finIm[segbase + tid] = xim;
        }
    }
}

// --------------------------------------------------------------------------
// Phase 2: serial combine of segment summaries -> carry-in per segment.
// carry[0] = 0 ; carry[s] = A^SEG * carry[s-1] + final[s-1]
// --------------------------------------------------------------------------
__global__ void ssm_combine(const float* __restrict__ Pre,
                            const float* __restrict__ Pim,
                            const float* __restrict__ finRe,
                            const float* __restrict__ finIm,
                            float* __restrict__ carryRe,
                            float* __restrict__ carryIm) {
    int id = blockIdx.x * blockDim.x + threadIdx.x;
    if (id >= SSM_B * SSM_H * SSM_N) return;
    int b = id / (SSM_H * SSM_N);
    int h = (id / SSM_N) % SSM_H;
    int n = id % SSM_N;
    float pr = Pre[h * SSM_N + n];
    float pi = Pim[h * SSM_N + n];
    float cre = 0.f, cim = 0.f;
    for (int s = 0; s < SSM_SEGS; ++s) {
        int idx = ((b * SSM_H + h) * SSM_SEGS + s) * SSM_N + n;
        carryRe[idx] = cre;
        carryIm[idx] = cim;
        float fr = finRe[idx], fi = finIm[idx];
        float nre = pr * cre - pi * cim + fr;
        float nim = pr * cim + pi * cre + fi;
        cre = nre; cim = nim;
    }
}

// --------------------------------------------------------------------------
extern "C" void kernel_launch(void* const* d_in, const int* in_sizes, int n_in,
                              void* d_out, int out_size, void* d_ws, size_t ws_size,
                              hipStream_t stream) {
    const float* u   = (const float*)d_in[0];
    const float* Lre = (const float*)d_in[1];
    const float* Lim = (const float*)d_in[2];
    const float* Bre = (const float*)d_in[3];
    const float* Bim = (const float*)d_in[4];
    const float* Cre = (const float*)d_in[5];
    const float* Cim = (const float*)d_in[6];
    const float* ldt = (const float*)d_in[7];
    const float* Dv  = (const float*)d_in[8];
    float* y = (float*)d_out;

    // Workspace layout (bytes)
    char* ws = (char*)d_ws;
    float*    Are = (float*)(ws);                 //  512 f32
    float*    Aim = (float*)(ws + 2048);
    float*    Pre = (float*)(ws + 4096);          //  A^SEG
    float*    Pim = (float*)(ws + 6144);
    _Float16* Wb  = (_Float16*)(ws + 8192);       // 131072 f16 (256 KB)
    _Float16* Wc  = (_Float16*)(ws + 8192 + 262144);
    float* finRe   = (float*)(ws + 8192 + 2 * 262144);            // 256 KB
    float* finIm   = (float*)(ws + 8192 + 3 * 262144);
    float* carryRe = (float*)(ws + 8192 + 4 * 262144);
    float* carryIm = (float*)(ws + 8192 + 5 * 262144);

    const int grid = SSM_B * SSM_H * SSM_SEGS;    // 1024 workgroups

    ssm_precompA<<<dim3(2), dim3(256), 0, stream>>>(Lre, Lim, ldt,
                                                    Are, Aim, Pre, Pim);
    ssm_precompW<<<dim3(1024), dim3(256), 0, stream>>>(Lre, Lim, ldt,
                                                       Bre, Bim, Cre, Cim, Wb, Wc);
    ssm_seg_kernel<false><<<dim3(grid), dim3(256), 0, stream>>>(
        u, Dv, Are, Aim, Wb, Wc, nullptr, nullptr, finRe, finIm, nullptr);
    ssm_combine<<<dim3(32), dim3(256), 0, stream>>>(Pre, Pim, finRe, finIm,
                                                    carryRe, carryIm);
    ssm_seg_kernel<true><<<dim3(grid), dim3(256), 0, stream>>>(
        u, Dv, Are, Aim, Wb, Wc, carryRe, carryIm, nullptr, nullptr, y);
}
